// TransformerModel_34531537060190
// MI455X (gfx1250) — matrix operand order repaired and checked by reading, not compiled, not run on hardware
//
#include <hip/hip_runtime.h>

// ---------------------------------------------------------------------------
// Types for CDNA5 WMMA (wave32, v_wmma_f32_16x16x32_bf16)
// ---------------------------------------------------------------------------
typedef __bf16 bf16_t;
typedef __attribute__((ext_vector_type(16))) __bf16 v16bf;
typedef __attribute__((ext_vector_type(8)))  __bf16 v8bf;
typedef __attribute__((ext_vector_type(8)))  float  v8f;
typedef __attribute__((ext_vector_type(4)))  int    v4i_t;

static constexpr int Lc  = 6;
static constexpr int Bc  = 4;
static constexpr int Tc  = 1024;
static constexpr int Ec  = 768;
static constexpr int Hc  = 12;
static constexpr int Vv  = 32000;
static constexpr int FFc = 4 * Ec;       // 3072
static constexpr int Mr  = Bc * Tc;      // 4096 token rows
static constexpr int HS  = Ec / Hc;      // 64 head dim

// Async global->LDS DMA (gfx1250). Guarded so the file compiles either way.
#if defined(__has_builtin)
#if __has_builtin(__builtin_amdgcn_global_load_async_to_lds_b128)
#define USE_ASYNC_LDS 1
#endif
#endif
#ifndef USE_ASYNC_LDS
#define USE_ASYNC_LDS 0
#endif

// Address-space casts via integer round-trip (HIP idiom; generic<->ASn pointer
// casts are rejected by clang). AS3 pointers are the low 32 bits of the
// generic LDS address (ISA 10.2: flat-LDS truncates to addr[31:0]).
#define AS1_V4I(p) ((__attribute__((address_space(1))) v4i_t*)(unsigned long long)(p))
#define AS3_V4I(p) ((__attribute__((address_space(3))) v4i_t*)(unsigned int)(unsigned long long)(p))

// ---------------------------------------------------------------------------
// Fused fp32 -> bf16 convert + transpose: in[K][N] -> out[N][K]
// 32x32 tiles staged through LDS so both sides stay coalesced.
// ---------------------------------------------------------------------------
__global__ __launch_bounds__(256) void cvt_transpose_kernel(
    const float* __restrict__ in, bf16_t* __restrict__ out, int K, int N) {
  __shared__ float tile[32][33];
  const int l = blockIdx.z;
  const float* inp = in + (size_t)l * K * N;
  bf16_t* outp = out + (size_t)l * K * N;
  int nb = blockIdx.x * 32, kb = blockIdx.y * 32;
  int tx = threadIdx.x & 31, ty = threadIdx.x >> 5;  // 32 x 8
#pragma unroll
  for (int i = 0; i < 32; i += 8)
    tile[ty + i][tx] = inp[(size_t)(kb + ty + i) * N + nb + tx];
  __syncthreads();
#pragma unroll
  for (int i = 0; i < 32; i += 8)
    outp[(size_t)(nb + ty + i) * K + kb + tx] = (bf16_t)tile[tx][ty + i];
}

// ---------------------------------------------------------------------------
// Embedding: x[row][e] = tok_emb[idx[row]][e] + pos_emb[row % T][e]
// ---------------------------------------------------------------------------
__global__ __launch_bounds__(256) void embed_kernel(
    const int* __restrict__ idx, const float* __restrict__ tok,
    const float* __restrict__ pos, float* __restrict__ x) {
  int row = blockIdx.x;
  int t   = row % Tc;
  int id  = idx[row];
  const float* tr = tok + (size_t)id * Ec;
  const float* pr = pos + (size_t)t * Ec;
  float* xr = x + (size_t)row * Ec;
  for (int i = threadIdx.x; i < Ec; i += 256) xr[i] = tr[i] + pr[i];
}

// ---------------------------------------------------------------------------
// LayerNorm (biased var, eps 1e-5) -> bf16 output
// ---------------------------------------------------------------------------
__global__ __launch_bounds__(256) void ln_bf16_kernel(
    const float* __restrict__ x, const float* __restrict__ g,
    const float* __restrict__ bta, bf16_t* __restrict__ out, int E) {
  __shared__ float rs[8], rq[8];
  int row = blockIdx.x;
  const float* xr = x + (size_t)row * E;
  int lane = threadIdx.x & 31, w = threadIdx.x >> 5;
  float s = 0.f, q = 0.f;
  for (int i = threadIdx.x; i < E; i += 256) { float v = xr[i]; s += v; q += v * v; }
#pragma unroll
  for (int off = 16; off; off >>= 1) {
    s += __shfl_xor(s, off, 32);
    q += __shfl_xor(q, off, 32);
  }
  if (lane == 0) { rs[w] = s; rq[w] = q; }
  __syncthreads();
  s = rs[0] + rs[1] + rs[2] + rs[3] + rs[4] + rs[5] + rs[6] + rs[7];
  q = rq[0] + rq[1] + rq[2] + rq[3] + rq[4] + rq[5] + rq[6] + rq[7];
  float mean = s / E;
  float var  = q / E - mean * mean;
  float rstd = rsqrtf(var + 1e-5f);
  bf16_t* orow = out + (size_t)row * E;
  for (int i = threadIdx.x; i < E; i += 256)
    orow[i] = (bf16_t)((xr[i] - mean) * rstd * g[i] + bta[i]);
}

// ---------------------------------------------------------------------------
// bf16 WMMA GEMM: Y[M,N] = act(A[M,K] @ Wt[N][K]^T + bias) (+ residual)
// Weights pre-transposed so BOTH operand tiles are K-contiguous 16B rows.
// Double-buffered LDS pipeline on GLOBAL_LOAD_ASYNC_TO_LDS_B128: tile i+1's
// DMA is in flight while tile i's WMMAs execute; s_wait_asynccnt 3 retires
// only the older tile's transfers (3 DMAs/thread stay outstanding).
// Block tile 64x128, 8 waves (2x4), each wave 2x2 WMMA tiles, BK=32.
// LDS rows padded to 40 elems (80 B) to keep fragment loads 16B-aligned.
// ---------------------------------------------------------------------------
template <bool BIAS, bool RESID, bool RELU, bool OUTF, bool OUTB>
__global__ __launch_bounds__(256) void gemm_bf16_kernel(
    const bf16_t* __restrict__ A, const bf16_t* __restrict__ Wt,
    const float* __restrict__ bias, const float* __restrict__ resid,
    float* __restrict__ outf, bf16_t* __restrict__ outb,
    int M, int N, int K) {
  __shared__ bf16_t As[2][64 * 40];
  __shared__ bf16_t Bs[2][128 * 40];
  const int tid  = threadIdx.x;
  const int lane = tid & 31, w = tid >> 5;
  const int wm = w >> 2, wn = w & 3;         // 2 x 4 wave grid
  const int half = lane >> 4, l16 = lane & 15;
  const int bm0 = blockIdx.y * 64, bn0 = blockIdx.x * 128;

  const int arow = tid >> 2, acol = (tid & 3) * 8;   // A: 8 elems/thread
  const int brow = tid >> 1, bcol = (tid & 1) * 16;  // B: 16 elems/thread

  v8f acc[2][2] = {};

  // one 16x16x32 WMMA step per (tm, tn) pair, fragments from LDS buffer `buf`
  auto compute_tile = [&](int buf) {
    v16bf af[2], bwf[2];
#pragma unroll
    for (int tm = 0; tm < 2; ++tm) {
      int m  = wm * 32 + tm * 16 + l16;
      int kb = half * 8;
      v8bf lo = *(const v8bf*)(&As[buf][m * 40 + kb]);
      v8bf hi = *(const v8bf*)(&As[buf][m * 40 + 16 + kb]);
#pragma unroll
      for (int i = 0; i < 8; ++i) { af[tm][i] = lo[i]; af[tm][i + 8] = hi[i]; }
    }
#pragma unroll
    for (int tn = 0; tn < 2; ++tn) {
      int n  = wn * 32 + tn * 16 + l16;
      int kb = half * 8;
      v8bf lo = *(const v8bf*)(&Bs[buf][n * 40 + kb]);
      v8bf hi = *(const v8bf*)(&Bs[buf][n * 40 + 16 + kb]);
#pragma unroll
      for (int i = 0; i < 8; ++i) { bwf[tn][i] = lo[i]; bwf[tn][i + 8] = hi[i]; }
    }
#pragma unroll
    for (int tm = 0; tm < 2; ++tm)
#pragma unroll
      for (int tn = 0; tn < 2; ++tn)
        acc[tm][tn] = __builtin_amdgcn_wmma_f32_16x16x32_bf16(
            false, af[tm], false, bwf[tn], (short)0, acc[tm][tn], false, false);
  };

#if USE_ASYNC_LDS
  auto issue_tile = [&](int k0, int buf) {
    __builtin_amdgcn_global_load_async_to_lds_b128(
        AS1_V4I(A + (size_t)(bm0 + arow) * K + (k0 + acol)),
        AS3_V4I(&As[buf][arow * 40 + acol]), 0, 0);
    __builtin_amdgcn_global_load_async_to_lds_b128(
        AS1_V4I(Wt + (size_t)(bn0 + brow) * K + (k0 + bcol)),
        AS3_V4I(&Bs[buf][brow * 40 + bcol]), 0, 0);
    __builtin_amdgcn_global_load_async_to_lds_b128(
        AS1_V4I(Wt + (size_t)(bn0 + brow) * K + (k0 + bcol + 8)),
        AS3_V4I(&Bs[buf][brow * 40 + bcol + 8]), 0, 0);
  };
  const int nIter = K >> 5;
  issue_tile(0, 0);
  for (int i = 0; i < nIter; ++i) {
    const int cur = i & 1;
    __syncthreads();  // buf[cur^1]'s readers (iteration i-1) are done
    if (i + 1 < nIter) {
      issue_tile((i + 1) << 5, cur ^ 1);               // overlap with compute
      asm volatile("s_wait_asynccnt 0x3" ::: "memory"); // only tile i must land
    } else {
      asm volatile("s_wait_asynccnt 0x0" ::: "memory");
    }
    __syncthreads();  // tile i visible block-wide
    compute_tile(cur);
  }
#else
  for (int k0 = 0; k0 < K; k0 += 32) {
    uint4 av  = *(const uint4*)(A  + (size_t)(bm0 + arow) * K + (k0 + acol));
    uint4 bv0 = *(const uint4*)(Wt + (size_t)(bn0 + brow) * K + (k0 + bcol));
    uint4 bv1 = *(const uint4*)(Wt + (size_t)(bn0 + brow) * K + (k0 + bcol + 8));
    __syncthreads();
    *(uint4*)(&As[0][arow * 40 + acol])     = av;
    *(uint4*)(&Bs[0][brow * 40 + bcol])     = bv0;
    *(uint4*)(&Bs[0][brow * 40 + bcol + 8]) = bv1;
    __syncthreads();
    compute_tile(0);
  }
#endif

  // Branch-free epilogue: batch residual loads, hoist bias, coalesced stores.
#pragma unroll
  for (int tm = 0; tm < 2; ++tm)
#pragma unroll
    for (int tn = 0; tn < 2; ++tn) {
      const int m0 = bm0 + wm * 32 + tm * 16 + 8 * half;
      const int n  = bn0 + wn * 32 + tn * 16 + l16;
      float bval = 0.0f;
      if constexpr (BIAS) bval = bias[n];
      float res[8];
      if constexpr (RESID) {
#pragma unroll
        for (int r = 0; r < 8; ++r) res[r] = resid[(size_t)(m0 + r) * N + n];
      }
#pragma unroll
      for (int r = 0; r < 8; ++r) {
        float v = acc[tm][tn][r] + bval;
        if constexpr (RESID) v += res[r];
        if constexpr (RELU) v = fmaxf(v, 0.0f);
        if constexpr (OUTF) outf[(size_t)(m0 + r) * N + n] = v;
        if constexpr (OUTB) outb[(size_t)(m0 + r) * N + n] = (bf16_t)v;
      }
    }
}

// ---------------------------------------------------------------------------
// Flash attention, one wave per (b, h, 16-query tile). Causal, scale=1/8.
// Q.K^T and P.V both on WMMA; streaming softmax in f32.
// ---------------------------------------------------------------------------
__global__ __launch_bounds__(32) void attn_kernel(
    const bf16_t* __restrict__ Q, const bf16_t* __restrict__ Kt,
    const bf16_t* __restrict__ Vt, bf16_t* __restrict__ O) {
  __shared__ bf16_t P[16 * 40];
  const int t0 = blockIdx.x * 16;
  const int h  = blockIdx.y;
  const int b  = blockIdx.z;
  const int lane = threadIdx.x;
  const int half = lane >> 4, l16 = lane & 15;
  const float scale = 0.125f;  // 1/sqrt(64)

  // Q A-fragments for dim chunks [0,32) and [32,64)
  v16bf aq[2];
  {
    int t = t0 + l16;
    const bf16_t* qrow = Q + ((size_t)(b * Tc + t) * Ec + h * HS);
    int kb = half * 8;
#pragma unroll
    for (int kk = 0; kk < 2; ++kk) {
      v8bf lo = *(const v8bf*)(qrow + kk * 32 + kb);
      v8bf hi = *(const v8bf*)(qrow + kk * 32 + 16 + kb);
#pragma unroll
      for (int i = 0; i < 8; ++i) { aq[kk][i] = lo[i]; aq[kk][i + 8] = hi[i]; }
    }
  }

  float Mrow[8], Srow[8];
  v8f   Oacc[4] = {};
#pragma unroll
  for (int r = 0; r < 8; ++r) { Mrow[r] = -1e30f; Srow[r] = 0.f; }

  for (int jb = 0; jb <= t0 + 15; jb += 32) {
    float sraw[2][8];
#pragma unroll
    for (int st = 0; st < 2; ++st) {
      v8f s8 = {};
      int j = jb + st * 16 + l16;  // key column for this lane
#pragma unroll
      for (int kk = 0; kk < 2; ++kk) {
        const bf16_t* krow = Kt + ((size_t)(b * Tc + j) * Ec + h * HS + kk * 32 + half * 8);
        v8bf lo = *(const v8bf*)krow;
        v8bf hi = *(const v8bf*)(krow + 16);
        v16bf bk;
#pragma unroll
        for (int i = 0; i < 8; ++i) { bk[i] = lo[i]; bk[i + 8] = hi[i]; }
        s8 = __builtin_amdgcn_wmma_f32_16x16x32_bf16(
            false, aq[kk], false, bk, (short)0, s8, false, false);
      }
#pragma unroll
      for (int r = 0; r < 8; ++r) {
        int m = r + 8 * half;
        float s = s8[r] * scale;
        sraw[st][r] = (j <= t0 + m) ? s : -1e30f;  // causal mask
      }
    }
    // streaming softmax update (reduce across the 16 lanes of each half)
    float alpha[8];
#pragma unroll
    for (int r = 0; r < 8; ++r) {
      float tmax = fmaxf(sraw[0][r], sraw[1][r]);
#pragma unroll
      for (int off = 1; off < 16; off <<= 1) tmax = fmaxf(tmax, __shfl_xor(tmax, off, 16));
      float Mn = fmaxf(Mrow[r], tmax);
      alpha[r] = __expf(Mrow[r] - Mn);
      float p0 = __expf(sraw[0][r] - Mn);
      float p1 = __expf(sraw[1][r] - Mn);
      float rs = p0 + p1;
#pragma unroll
      for (int off = 1; off < 16; off <<= 1) rs += __shfl_xor(rs, off, 16);
      Srow[r] = Srow[r] * alpha[r] + rs;
      Mrow[r] = Mn;
      int m = r + 8 * half;
      P[m * 40 + l16]      = (bf16_t)p0;
      P[m * 40 + 16 + l16] = (bf16_t)p1;
    }
#pragma unroll
    for (int g = 0; g < 4; ++g)
#pragma unroll
      for (int r = 0; r < 8; ++r) Oacc[g][r] *= alpha[r];

    asm volatile("s_wait_dscnt 0" ::: "memory");  // LDS P stores visible

    // P as A-fragment (16 queries x 32 keys)
    v16bf pa;
    {
      int kb = half * 8;
      v8bf lo = *(const v8bf*)(&P[l16 * 40 + kb]);
      v8bf hi = *(const v8bf*)(&P[l16 * 40 + 16 + kb]);
#pragma unroll
      for (int i = 0; i < 8; ++i) { pa[i] = lo[i]; pa[i + 8] = hi[i]; }
    }
    // O += P @ V  (4 dim groups of 16)
#pragma unroll
    for (int g = 0; g < 4; ++g) {
      v16bf bv;
#pragma unroll
      for (int i = 0; i < 16; ++i) {
        int j = jb + half * 8 + (i & 7) + (i >> 3) * 16;
        bv[i] = Vt[(size_t)(b * Tc + j) * Ec + h * HS + g * 16 + l16];
      }
      Oacc[g] = __builtin_amdgcn_wmma_f32_16x16x32_bf16(
          false, pa, false, bv, (short)0, Oacc[g], false, false);
    }
  }

  // finalize: divide by row sum, write bf16
#pragma unroll
  for (int r = 0; r < 8; ++r) {
    int m = r + 8 * half;
    int t = t0 + m;
    float inv = 1.0f / Srow[r];
#pragma unroll
    for (int g = 0; g < 4; ++g)
      O[(size_t)(b * Tc + t) * Ec + h * HS + g * 16 + l16] = (bf16_t)(Oacc[g][r] * inv);
  }
}

// ---------------------------------------------------------------------------
// Loss: per-row logsumexp - target logit, atomicAdd mean into *loss
// ---------------------------------------------------------------------------
__global__ void zero_kernel(float* p) { if (threadIdx.x == 0 && blockIdx.x == 0) *p = 0.f; }

__global__ __launch_bounds__(256) void loss_kernel(
    const float* __restrict__ logits, const int* __restrict__ targets,
    float* __restrict__ loss, int V, float sc) {
  __shared__ float red[8];
  int row = blockIdx.x;
  const float* lr = logits + (size_t)row * V;
  int lane = threadIdx.x & 31, w = threadIdx.x >> 5;
  float mx = -1e30f;
  for (int i = threadIdx.x; i < V; i += 256) mx = fmaxf(mx, lr[i]);
#pragma unroll
  for (int off = 16; off; off >>= 1) mx = fmaxf(mx, __shfl_xor(mx, off, 32));
  if (lane == 0) red[w] = mx;
  __syncthreads();
  mx = fmaxf(fmaxf(fmaxf(red[0], red[1]), fmaxf(red[2], red[3])),
             fmaxf(fmaxf(red[4], red[5]), fmaxf(red[6], red[7])));
  __syncthreads();
  float se = 0.f;
  for (int i = threadIdx.x; i < V; i += 256) se += __expf(lr[i] - mx);
#pragma unroll
  for (int off = 16; off; off >>= 1) se += __shfl_xor(se, off, 32);
  if (lane == 0) red[w] = se;
  __syncthreads();
  if (threadIdx.x == 0) {
    float s = red[0] + red[1] + red[2] + red[3] + red[4] + red[5] + red[6] + red[7];
    float lse = mx + __logf(s);
    atomicAdd(loss, (lse - lr[targets[row]]) * sc);
  }
}

// ---------------------------------------------------------------------------
// Host orchestration
// ---------------------------------------------------------------------------
extern "C" void kernel_launch(void* const* d_in, const int* in_sizes, int n_in,
                              void* d_out, int out_size, void* d_ws, size_t ws_size,
                              hipStream_t stream) {
  const int*   idx     = (const int*)d_in[0];
  const int*   targets = (const int*)d_in[1];
  const float* tok_emb = (const float*)d_in[2];
  const float* pos_emb = (const float*)d_in[3];
  const float* Wq  = (const float*)d_in[4];
  const float* Wk  = (const float*)d_in[5];
  const float* Wv  = (const float*)d_in[6];
  const float* Wp  = (const float*)d_in[7];
  const float* bp  = (const float*)d_in[8];
  const float* W1  = (const float*)d_in[9];
  const float* b1  = (const float*)d_in[10];
  const float* W2  = (const float*)d_in[11];
  const float* b2  = (const float*)d_in[12];
  const float* ln1g = (const float*)d_in[13];
  const float* ln1b = (const float*)d_in[14];
  const float* ln2g = (const float*)d_in[15];
  const float* ln2b = (const float*)d_in[16];
  const float* lnfg = (const float*)d_in[17];
  const float* lnfb = (const float*)d_in[18];
  const float* Wlm = (const float*)d_in[19];
  const float* blm = (const float*)d_in[20];

  // workspace carve-out
  char* wp = (char*)d_ws;
  auto take = [&](size_t bytes) -> void* {
    void* r = (void*)wp;
    wp += (bytes + 255) & ~(size_t)255;
    return r;
  };
  float*  x    = (float*)take((size_t)Mr * Ec * 4);
  bf16_t* lnb  = (bf16_t*)take((size_t)Mr * Ec * 2);
  bf16_t* qb   = (bf16_t*)take((size_t)Mr * Ec * 2);
  bf16_t* kbuf = (bf16_t*)take((size_t)Mr * Ec * 2);
  bf16_t* vb   = (bf16_t*)take((size_t)Mr * Ec * 2);
  bf16_t* ab   = (bf16_t*)take((size_t)Mr * Ec * 2);
  bf16_t* hb   = (bf16_t*)take((size_t)Mr * FFc * 2);
  bf16_t* Wqb  = (bf16_t*)take((size_t)Lc * Ec * Ec * 2);   // transposed [N][K]
  bf16_t* Wkb  = (bf16_t*)take((size_t)Lc * Ec * Ec * 2);
  bf16_t* Wvb  = (bf16_t*)take((size_t)Lc * Ec * Ec * 2);
  bf16_t* Wpb  = (bf16_t*)take((size_t)Lc * Ec * Ec * 2);
  bf16_t* W1b  = (bf16_t*)take((size_t)Lc * Ec * FFc * 2);
  bf16_t* W2b  = (bf16_t*)take((size_t)Lc * FFc * Ec * 2);
  bf16_t* Wlmb = (bf16_t*)take((size_t)Ec * Vv * 2);

  float* logits = (float*)d_out;
  float* loss   = logits + (size_t)Mr * Vv;

  // transpose-convert: in [L][K][N] fp32 -> out [L][N][K] bf16
  auto cvt_t = [&](const float* in, bf16_t* out, int K, int N, int L) {
    dim3 grid(N / 32, K / 32, L);
    cvt_transpose_kernel<<<grid, 256, 0, stream>>>(in, out, K, N);
  };

  // GEMM variant launchers (weights pre-transposed: Wt[N][K])
  auto gemm_qkv = [&](const bf16_t* A, const bf16_t* W, bf16_t* outb,
                      int M, int N, int K) {
    dim3 grid(N / 128, M / 64);
    gemm_bf16_kernel<false, false, false, false, true>
        <<<grid, 256, 0, stream>>>(A, W, nullptr, nullptr, nullptr, outb, M, N, K);
  };
  auto gemm_res = [&](const bf16_t* A, const bf16_t* W, const float* bias,
                      const float* resid, float* outf, int M, int N, int K) {
    dim3 grid(N / 128, M / 64);
    gemm_bf16_kernel<true, true, false, true, false>
        <<<grid, 256, 0, stream>>>(A, W, bias, resid, outf, nullptr, M, N, K);
  };
  auto gemm_relu = [&](const bf16_t* A, const bf16_t* W, const float* bias,
                       bf16_t* outb, int M, int N, int K) {
    dim3 grid(N / 128, M / 64);
    gemm_bf16_kernel<true, false, true, false, true>
        <<<grid, 256, 0, stream>>>(A, W, bias, nullptr, nullptr, outb, M, N, K);
  };
  auto gemm_lm = [&](const bf16_t* A, const bf16_t* W, const float* bias,
                     float* outf, int M, int N, int K) {
    dim3 grid(N / 128, M / 64);
    gemm_bf16_kernel<true, false, false, true, false>
        <<<grid, 256, 0, stream>>>(A, W, bias, nullptr, outf, nullptr, M, N, K);
  };

  // 1) convert + transpose weights to bf16 (per call; no caching)
  cvt_t(Wq,  Wqb,  Ec,  Ec,  Lc);
  cvt_t(Wk,  Wkb,  Ec,  Ec,  Lc);
  cvt_t(Wv,  Wvb,  Ec,  Ec,  Lc);
  cvt_t(Wp,  Wpb,  Ec,  Ec,  Lc);
  cvt_t(W1,  W1b,  Ec,  FFc, Lc);
  cvt_t(W2,  W2b,  FFc, Ec,  Lc);
  cvt_t(Wlm, Wlmb, Ec,  Vv,  1);

  // 2) embedding
  embed_kernel<<<Mr, 256, 0, stream>>>(idx, tok_emb, pos_emb, x);

  // 3) transformer layers
  for (int l = 0; l < Lc; ++l) {
    const size_t we = (size_t)l * Ec * Ec;
    const size_t wf = (size_t)l * Ec * FFc;
    ln_bf16_kernel<<<Mr, 256, 0, stream>>>(x, ln1g + (size_t)l * Ec, ln1b + (size_t)l * Ec, lnb, Ec);
    gemm_qkv(lnb, Wqb + we, qb,   Mr, Ec, Ec);
    gemm_qkv(lnb, Wkb + we, kbuf, Mr, Ec, Ec);
    gemm_qkv(lnb, Wvb + we, vb,   Mr, Ec, Ec);
    attn_kernel<<<dim3(Tc / 16, Hc, Bc), 32, 0, stream>>>(qb, kbuf, vb, ab);
    gemm_res(ab, Wpb + we, bp + (size_t)l * Ec, x, x, Mr, Ec, Ec);
    ln_bf16_kernel<<<Mr, 256, 0, stream>>>(x, ln2g + (size_t)l * Ec, ln2b + (size_t)l * Ec, lnb, Ec);
    gemm_relu(lnb, W1b + wf, b1 + (size_t)l * FFc, hb, Mr, FFc, Ec);
    gemm_res(hb,  W2b + wf, b2 + (size_t)l * Ec, x, x, Mr, Ec, FFc);
  }

  // 4) final LN + LM head -> logits (fp32, directly into d_out)
  ln_bf16_kernel<<<Mr, 256, 0, stream>>>(x, lnfg, lnfb, lnb, Ec);
  gemm_lm(lnb, Wlmb, blm, logits, Mr, Vv, Ec);

  // 5) loss = mean(logsumexp(row) - logit[target])
  zero_kernel<<<1, 32, 0, stream>>>(loss);
  loss_kernel<<<Mr, 256, 0, stream>>>(logits, targets, loss, Vv, 1.0f / (float)Mr);
}
